// MultiGeometryGPT_9964324127136
// MI455X (gfx1250) — matrix operand order repaired and checked by reading, not compiled, not run on hardware
//
#include <hip/hip_runtime.h>
#include <hip/hip_bf16.h>

// ---------------------------------------------------------------------------
// MultiGeometryGPT block for MI455X (gfx1250, wave32, WMMA bf16 16x16x32).
// Fragment-layout LDS everywhere (subtile = 32 lanes x 8 u32 of bf16 pairs;
// fragment load = one 32B v16bf load = 2x ds_load_b128).  Weights and
// row-produced activations pre-swizzled in global; GEMM staging is
// double-buffered global_load_async_to_lds_b128 with partial asynccnt waits.
// ---------------------------------------------------------------------------
typedef __bf16 bf16;
typedef __attribute__((ext_vector_type(16))) __bf16 v16bf;
typedef __attribute__((ext_vector_type(8)))  float  v8f;
typedef int vi4 __attribute__((vector_size(16)));
typedef __attribute__((address_space(1))) vi4* gvi4p;   // global int4*
typedef __attribute__((address_space(3))) vi4* lvi4p;   // LDS int4*

#define BB 2
#define TT 2048
#define EE 1024
#define HH 16
#define DD 64
#define MM (BB * TT)            // 4096 rows (always multiple of 128)

#if __has_builtin(__builtin_amdgcn_global_load_async_to_lds_b128) && \
    __has_builtin(__builtin_amdgcn_s_wait_asynccnt)
#define HAVE_ASYNC 1
#else
#define HAVE_ASYNC 0
#endif

__device__ inline v8f v8f_zero() {
  v8f z;
#pragma unroll
  for (int i = 0; i < 8; ++i) z[i] = 0.f;
  return z;
}

// pack two fp32 -> packed bf16 pair (lo = even k, hi = odd k)
__device__ inline unsigned pk2(float a, float b) {
  unsigned short ua = __builtin_bit_cast(unsigned short, (bf16)a);
  unsigned short ub = __builtin_bit_cast(unsigned short, (bf16)b);
  return ((unsigned)ub << 16) | ua;
}

__device__ inline v16bf ldfrag(const unsigned* p) {
  return *(const v16bf*)p;
}

// contiguous 2048-u32 global -> LDS tile copy (8 subtiles), async when avail
__device__ inline void tile_copy(const unsigned* src, unsigned* dst, int tid) {
#if HAVE_ASYNC
#pragma unroll
  for (int it = 0; it < 2; ++it) {
    int e = tid + it * 256;   // 512 x 16B
    __builtin_amdgcn_global_load_async_to_lds_b128(
        (gvi4p)(src + e * 4), (lvi4p)(dst + e * 4), 0, 0);
  }
#else
#pragma unroll
  for (int it = 0; it < 2; ++it) {
    int e = tid + it * 256;
    ((uint4*)dst)[e] = ((const uint4*)src)[e];
  }
#endif
}

template <int N_>
__device__ inline void wait_async_c() {
#if HAVE_ASYNC
  __builtin_amdgcn_s_wait_asynccnt(N_);
#endif
}

__device__ inline float block_reduce_sum(float v, float* red) {
  int tid = threadIdx.x;
  red[tid] = v;
  __syncthreads();
#pragma unroll
  for (int off = 128; off > 0; off >>= 1) {
    if (tid < off) red[tid] += red[tid + off];
    __syncthreads();
  }
  float r = red[0];
  __syncthreads();
  return r;
}

// ---------------------------------------------------------------------------
// Weight pre-swizzle: W[K,N] fp32 -> Wz B-frag tiles
// Wz[((kt*NT + nt)*32 + lane)*8 + slot]; lane=((k>>3)&1)*16+(n&15),
// slot=((k>>4)&1)*4+((k&7)>>1), pair=(k,k+1), zero-padded tails.
// ---------------------------------------------------------------------------
__global__ void __launch_bounds__(256) wswz_kernel(
    const float* __restrict__ W, unsigned* __restrict__ Wz,
    int K, int N, int NT, int KT) {
  size_t idx = (size_t)blockIdx.x * 256 + threadIdx.x;
  size_t total = (size_t)KT * NT * 256;
  if (idx >= total) return;
  int slot = (int)(idx & 7);
  int lane = (int)((idx >> 3) & 31);
  size_t tile = idx >> 8;
  int nt = (int)(tile % NT);
  int kt = (int)(tile / NT);
  int k = kt * 32 + (slot >> 2) * 16 + (lane >> 4) * 8 + (slot & 3) * 2;
  int n = nt * 16 + (lane & 15);
  float b0 = (k < K && n < N) ? W[(size_t)k * N + n] : 0.f;
  float b1 = (k + 1 < K && n < N) ? W[(size_t)(k + 1) * N + n] : 0.f;
  Wz[idx] = pk2(b0, b1);
}

// ---------------------------------------------------------------------------
// row_project: row -> [x0, y] with x0 = sqrt(exp(curv)+|y|^2),
// y = (optional LN of) src row.  Emits optional fp32 row (dst) and optional
// A-frag swizzled bf16 row (Az, layout [kt][mt][lane*8+slot], MT=grid/16).
// n == 1024 here; padded to 1056 (33 k-tiles).
// ---------------------------------------------------------------------------
__global__ void __launch_bounds__(256) row_project_kernel(
    const float* __restrict__ src, float* __restrict__ dst,
    unsigned* __restrict__ Az,
    const float* __restrict__ lw, const float* __restrict__ lb,
    const float* __restrict__ curv, int n, int do_ln) {
  int row = blockIdx.x, tid = threadIdx.x;
  const float* x = src + (size_t)row * n;
  __shared__ float red[256];
  __shared__ float ybuf[1056];

  float xv[4];
  int cnt = (n + 255) >> 8;
  float mean = 0.f, inv_std = 1.f;
#pragma unroll
  for (int i = 0; i < 4; ++i) {
    int e = tid + 256 * i;
    xv[i] = (i < cnt && e < n) ? x[e] : 0.f;
  }
  if (do_ln) {
    float s = 0.f;
#pragma unroll
    for (int i = 0; i < 4; ++i) s += xv[i];
    mean = block_reduce_sum(s, red) / (float)n;
    float s2 = 0.f;
#pragma unroll
    for (int i = 0; i < 4; ++i) {
      int e = tid + 256 * i;
      if (i < cnt && e < n) { float d = xv[i] - mean; s2 += d * d; }
    }
    float var = block_reduce_sum(s2, red) / (float)n;
    inv_std = rsqrtf(var + 1e-5f);
  }
  float ss = 0.f;
#pragma unroll
  for (int i = 0; i < 4; ++i) {
    int e = tid + 256 * i;
    if (i < cnt && e < n) {
      float v = do_ln ? ((xv[i] - mean) * inv_std * lw[e] + lb[e]) : xv[i];
      ybuf[1 + e] = v;
      if (dst) dst[(size_t)row * (n + 1) + 1 + e] = v;
      ss += v * v;
    }
  }
  // zero pad cols n+1 .. 1055
  for (int e = n + 1 + tid; e < 1056; e += 256) ybuf[e] = 0.f;
  float tot = block_reduce_sum(ss, red);
  if (tid == 0) {
    float x0 = sqrtf(__expf(curv[0]) + tot);
    ybuf[0] = x0;
    if (dst) dst[(size_t)row * (n + 1)] = x0;
  }
  __syncthreads();
  if (Az) {
    int mt = row >> 4, rl = row & 15;
    int MT = (int)(gridDim.x >> 4);
    for (int p = tid; p < 528; p += 256) {       // 1056/2 pairs
      int kt = p >> 4, pp = p & 15;
      int lt = ((pp >> 2) & 1) * 16 + rl;
      int sl = ((pp >> 3) << 2) | (pp & 3);
      Az[((size_t)kt * MT + mt) * 256 + lt * 8 + sl] =
          pk2(ybuf[2 * p], ybuf[2 * p + 1]);
    }
  }
}

// ---------------------------------------------------------------------------
// gelu_project: row of g[n=4099] -> swizzled A-frag row of [x0, gelu(g)]
// (4100 cols padded to 4128 = 129 k-tiles).  x0 = sqrt(exp(curv)+|gelu|^2).
// ---------------------------------------------------------------------------
__global__ void __launch_bounds__(256) gelu_project_kernel(
    const float* __restrict__ src, unsigned* __restrict__ Az,
    const float* __restrict__ curv, int n) {
  int row = blockIdx.x, tid = threadIdx.x;
  const float* x = src + (size_t)row * n;
  __shared__ float red[256];
  __shared__ float ybuf[4128];
  float ss = 0.f;
  for (int e = tid; e < n; e += 256) {
    float v = x[e];
    float g = 0.5f * v * (1.0f + erff(v * 0.70710678118654752f));
    ybuf[1 + e] = g;
    ss += g * g;
  }
  for (int e = n + 1 + tid; e < 4128; e += 256) ybuf[e] = 0.f;
  float tot = block_reduce_sum(ss, red);
  if (tid == 0) ybuf[0] = sqrtf(__expf(curv[0]) + tot);
  __syncthreads();
  int mt = row >> 4, rl = row & 15;
  int MT = (int)(gridDim.x >> 4);
  for (int p = tid; p < 2064; p += 256) {        // 4128/2 pairs
    int kt = p >> 4, pp = p & 15;
    int lt = ((pp >> 2) & 1) * 16 + rl;
    int sl = ((pp >> 3) << 2) | (pp & 3);
    Az[((size_t)kt * MT + mt) * 256 + lt * 8 + sl] =
        pk2(ybuf[2 * p], ybuf[2 * p + 1]);
  }
}

// ---------------------------------------------------------------------------
// WMMA GEMM: C[M,N] = A @ W (+bias) (+resid).  W pre-swizzled (Wz).
// A either pre-swizzled (Az) or fp32 (converted in-kernel, attn_proj only).
// Double-buffered LDS with partial s_wait_asynccnt pipelining.
// 256 thr = 8 waves, block tile 128x128x32, wave tile 32x64 (2x4 of 16x16).
// M must be a multiple of 128.
// ---------------------------------------------------------------------------
__global__ void __launch_bounds__(256) gemm_wmma_kernel(
    const float* __restrict__ A, const unsigned* __restrict__ Az,
    const unsigned* __restrict__ Wz,
    const float* __restrict__ bias, const float* __restrict__ resid,
    float* __restrict__ C, int M, int N, int K, int NT) {
  __shared__ alignas(32) unsigned As[2][8][256];   // double-buffered
  __shared__ alignas(32) unsigned Bs[2][8][256];

  int tid = threadIdx.x;
  int lane = tid & 31, wave = tid >> 5;
  int half = lane >> 4, col = lane & 15;
  int bm = blockIdx.y * 128, bn = blockIdx.x * 128;
  int mt0 = (wave >> 1) * 2;       // 2 m-subtiles per wave
  int nt0 = (wave & 1) * 4;        // 4 n-subtiles per wave
  int KT = (K + 31) >> 5;
  int MT = M >> 4;

  v8f acc[2][4];
#pragma unroll
  for (int i = 0; i < 2; ++i)
#pragma unroll
    for (int j = 0; j < 4; ++j) acc[i][j] = v8f_zero();

  auto stage = [&](int kts, unsigned (*Asb)[256], unsigned (*Bsb)[256]) {
    const unsigned* bsrc = Wz + ((size_t)kts * NT + (bn >> 4)) * 256;
    tile_copy(bsrc, &Bsb[0][0], tid);
    if (Az) {
      const unsigned* asrc = Az + ((size_t)kts * MT + (bm >> 4)) * 256;
      tile_copy(asrc, &Asb[0][0], tid);
    } else {
      int k0 = kts * 32;
      if (k0 + 32 <= K) {
        // full tile: no guards
#pragma unroll
        for (int it = 0; it < 8; ++it) {
          int e = tid + it * 256;
          int r = e >> 4, p = e & 15;
          const float* ap = A + (size_t)(bm + r) * K + k0 + 2 * p;
          int lt = ((p >> 2) & 1) * 16 + (r & 15);
          int sl = ((p >> 3) << 2) | (p & 3);
          Asb[r >> 4][lt * 8 + sl] = pk2(ap[0], ap[1]);
        }
      } else {
        // tail tile: branch-free clamped loads with zero masks
#pragma unroll
        for (int it = 0; it < 8; ++it) {
          int e = tid + it * 256;
          int r = e >> 4, p = e & 15;
          int gk = k0 + 2 * p;
          int gk0 = gk < K - 1 ? gk : K - 1;
          const float* ap = A + (size_t)(bm + r) * K;
          float m0 = gk < K ? 1.f : 0.f;
          float m1 = gk + 1 < K ? 1.f : 0.f;
          float a0 = ap[gk0] * m0;
          float a1 = ap[gk0 + 1] * m1;
          int lt = ((p >> 2) & 1) * 16 + (r & 15);
          int sl = ((p >> 3) << 2) | (p & 3);
          Asb[r >> 4][lt * 8 + sl] = pk2(a0, a1);
        }
      }
    }
  };

  stage(0, As[0], Bs[0]);
  for (int kt = 0; kt < KT; ++kt) {
    int buf = kt & 1;
    if (kt + 1 < KT) {
      if (kt + 2 < KT)
        __builtin_prefetch(Wz + ((size_t)(kt + 2) * NT + (bn >> 4)) * 256, 0, 1);
      stage(kt + 1, As[buf ^ 1], Bs[buf ^ 1]);
      // async loads complete in order: leaving the newly issued group
      // outstanding guarantees tile kt's copies have landed.
      if (Az) wait_async_c<4>(); else wait_async_c<2>();
    } else {
      wait_async_c<0>();
    }
    __syncthreads();

    v16bf af[2], bfr[4];
#pragma unroll
    for (int i = 0; i < 2; ++i) af[i] = ldfrag(&As[buf][mt0 + i][lane * 8]);
#pragma unroll
    for (int j = 0; j < 4; ++j) bfr[j] = ldfrag(&Bs[buf][nt0 + j][lane * 8]);
#pragma unroll
    for (int i = 0; i < 2; ++i)
#pragma unroll
      for (int j = 0; j < 4; ++j)
        acc[i][j] = __builtin_amdgcn_wmma_f32_16x16x32_bf16(
            false, af[i], false, bfr[j], (short)0, acc[i][j], false, false);
  }

#pragma unroll
  for (int i = 0; i < 2; ++i)
#pragma unroll
    for (int j = 0; j < 4; ++j) {
      int gn = bn + (nt0 + j) * 16 + col;
      if (gn < N) {
#pragma unroll
        for (int r = 0; r < 8; ++r) {
          int gm = bm + (mt0 + i) * 16 + r + 8 * half;
          float v = acc[i][j][r];
          if (bias)  v += bias[gn];
          if (resid) v += resid[(size_t)gm * N + gn];
          C[(size_t)gm * N + gn] = v;
        }
      }
    }
}

// ---------------------------------------------------------------------------
// RoPE + Lorentz time components.  row = (b*H + h)*T + t, 64 threads/row.
// ---------------------------------------------------------------------------
__global__ void __launch_bounds__(256) rope_kernel(
    const float* __restrict__ qkv, const float* __restrict__ curv,
    float* __restrict__ Qr, float* __restrict__ Kr, float* __restrict__ Vr,
    float* __restrict__ qt, float* __restrict__ kt, float* __restrict__ lv0) {
  int sub = threadIdx.x >> 6;
  int d   = threadIdx.x & 63;
  int row = blockIdx.x * 4 + sub;
  int t = row & (TT - 1);
  int h = (row >> 11) & (HH - 1);
  int b = row >> 15;
  __shared__ float red[256];

  const float* base = qkv + ((size_t)b * TT + t) * (3 * EE);
  int j = d & 31;
  float q1 = base[h * DD + j];
  float q2 = base[h * DD + 32 + j];
  float k1 = base[EE + h * DD + j];
  float k2 = base[EE + h * DD + 32 + j];
  float v  = base[2 * EE + h * DD + d];

  float inv = __expf(-((float)(2 * j) / 64.f) * 9.210340371976184f);
  float ang = (float)t * inv;
  float c = __cosf(ang), s = __sinf(ang);
  float qr = (d < 32) ? (q1 * c + q2 * s) : (-q1 * s + q2 * c);
  float kr = (d < 32) ? (k1 * c + k2 * s) : (-k1 * s + k2 * c);

  float sq, sk, sv;
  red[threadIdx.x] = qr * qr; __syncthreads();
#pragma unroll
  for (int off = 32; off > 0; off >>= 1) {
    if (d < off) red[threadIdx.x] += red[threadIdx.x + off];
    __syncthreads();
  }
  sq = red[sub * 64]; __syncthreads();
  red[threadIdx.x] = kr * kr; __syncthreads();
#pragma unroll
  for (int off = 32; off > 0; off >>= 1) {
    if (d < off) red[threadIdx.x] += red[threadIdx.x + off];
    __syncthreads();
  }
  sk = red[sub * 64]; __syncthreads();
  red[threadIdx.x] = v * v; __syncthreads();
#pragma unroll
  for (int off = 32; off > 0; off >>= 1) {
    if (d < off) red[threadIdx.x] += red[threadIdx.x + off];
    __syncthreads();
  }
  sv = red[sub * 64]; __syncthreads();

  float Kh = __expf(curv[h]);
  size_t o = (size_t)row * DD + d;
  Qr[o] = qr; Kr[o] = kr; Vr[o] = v;
  if (d == 0) {
    qt[row]  = sqrtf(Kh + sq);
    kt[row]  = sqrtf(Kh + sk);
    lv0[row] = sqrtf(Kh + sv);
  }
}

// ---------------------------------------------------------------------------
// Hyperbolic flash attention, fragment-layout LDS everywhere.
// grid = (T/64, B*H), 256 thr (8 waves). Wave: 1 m-subtile (16 q rows),
// 2 key-subtiles for scores, 3 e-subtiles (of 6, e padded to 96) for P@LV.
// ---------------------------------------------------------------------------
__global__ void __launch_bounds__(256) attn_kernel(
    const float* __restrict__ Qr, const float* __restrict__ Kr,
    const float* __restrict__ Vr, const float* __restrict__ qtG,
    const float* __restrict__ ktG, const float* __restrict__ lv0G,
    const float* __restrict__ curv, float* __restrict__ O) {
  int bh = blockIdx.y;
  int h = bh & (HH - 1);
  int b = bh >> 4;
  int qb = blockIdx.x;
  int q0 = qb * 64;
  int tid = threadIdx.x, lane = tid & 31, wave = tid >> 5;
  int half = lane >> 4, col = lane & 15;
  int mt  = wave >> 1;           // 0..3 (16 q rows each)
  int ns0 = (wave & 1) * 2;      // key subtile base (of 4)
  int no0 = (wave & 1) * 3;      // e   subtile base (of 6)

  float Kh = __expf(curv[h]);
  float sqrtKh = sqrtf(Kh);

  __shared__ alignas(32) unsigned Qf[4][2][256];   // [mt][kt(d)]    8KB
  __shared__ alignas(32) unsigned Kf[2][4][256];   // [kt(d)][nt]    8KB
  __shared__ alignas(32) unsigned LVf[2][6][256];  // [kt(s)][nt(e)] 12KB
  __shared__ float qtS[64], ktS[64], rowm[64], rowl[64], rowscale[64];
  __shared__ union {
    struct { float S[64][68]; alignas(32) unsigned Pf[4][2][256]; } sp;
    float Ost[64][97];
  } u;

  const size_t bhT = (size_t)bh * TT;

  // ---- stage Q (A-frag layout, pairs over d) ----
  const float2* Q2 = (const float2*)(Qr + (bhT + q0) * DD);
#pragma unroll
  for (int it = 0; it < 8; ++it) {
    int e = tid + it * 256;              // e < 2048
    int r = e >> 5, p = e & 31;          // q-row, d-pair
    float2 qv = Q2[r * 32 + p];
    int ktl = p >> 4, pp = p & 15;
    int lt = ((pp >> 2) & 1) * 16 + (r & 15);
    int sl = ((pp >> 3) << 2) | (pp & 3);
    Qf[r >> 4][ktl][lt * 8 + sl] = pk2(qv.x, qv.y);
  }
  // zero-fill LV pad region (e=65..95) once; e=64 column is rewritten per
  // key block, e>=65 never touched again.
  {
    unsigned* z0 = &LVf[0][4][0];
    unsigned* z1 = &LVf[1][4][0];
#pragma unroll
    for (int it = 0; it < 2; ++it) {
      int e = tid + it * 256;
      z0[e] = 0u; z1[e] = 0u;
    }
  }
  if (tid < 64) {
    qtS[tid]  = qtG[bhT + q0 + tid];
    rowm[tid] = -3.0e38f;
    rowl[tid] = 0.f;
  }
  __syncthreads();

  v8f o0 = v8f_zero(), o1 = v8f_zero(), o2 = v8f_zero();

  for (int kb = 0; kb <= qb; ++kb) {
    int s0 = kb * 64;
    // ---- stage K (B-frag layout over [d][key]) ----
    const float2* K2 = (const float2*)(Kr + (bhT + s0) * DD);
#pragma unroll
    for (int it = 0; it < 8; ++it) {
      int e = tid + it * 256;
      int jj = e >> 5, p = e & 31;       // key, d-pair
      float2 kv = K2[jj * 32 + p];
      int ktl = p >> 4, pp = p & 15;
      int lt = ((pp >> 2) & 1) * 16 + (jj & 15);
      Kf[ktl][jj >> 4][lt * 8 + (((pp >> 3) << 2) | (pp & 3))] = pk2(kv.x, kv.y);
    }
    // ---- stage LV cols 1..64 from Vr (branch-free) ----
    const float* Vb = Vr + (bhT + s0) * DD;
#pragma unroll
    for (int it = 0; it < 8; ++it) {
      int e = tid + it * 256;            // e < 2048
      int p = e >> 6, cc = (e & 63) + 1; // s-pair, e-col 1..64
      float va = Vb[(2 * p) * DD + (e & 63)];
      float vb = Vb[(2 * p + 1) * DD + (e & 63)];
      int ktl = p >> 4, pp = p & 15;
      int lt = ((pp >> 2) & 1) * 16 + (cc & 15);
      LVf[ktl][cc >> 4][lt * 8 + (((pp >> 3) << 2) | (pp & 3))] = pk2(va, vb);
    }
    if (tid < 32) {                      // e-col 0 from lv0
      int p = tid;
      float va = lv0G[bhT + s0 + 2 * p];
      float vb = lv0G[bhT + s0 + 2 * p + 1];
      int ktl = p >> 4, pp = p & 15;
      int lt = ((pp >> 2) & 1) * 16;
      LVf[ktl][0][lt * 8 + (((pp >> 3) << 2) | (pp & 3))] = pk2(va, vb);
    }
    if (tid < 64) ktS[tid] = ktG[bhT + s0 + tid];
    __syncthreads();

    // ---- scores: 16 q-rows x 32 keys per wave ----
    v8f sacc[2]; sacc[0] = v8f_zero(); sacc[1] = v8f_zero();
#pragma unroll
    for (int ktl = 0; ktl < 2; ++ktl) {
      v16bf af = ldfrag(&Qf[mt][ktl][lane * 8]);
#pragma unroll
      for (int jt = 0; jt < 2; ++jt) {
        v16bf bfr = ldfrag(&Kf[ktl][ns0 + jt][lane * 8]);
        sacc[jt] = __builtin_amdgcn_wmma_f32_16x16x32_bf16(
            false, af, false, bfr, (short)0, sacc[jt], false, false);
      }
    }
    // hyperbolic distance transform + causal mask -> S
#pragma unroll
    for (int jt = 0; jt < 2; ++jt) {
      int kc = (ns0 + jt) * 16 + col;
      int key = s0 + kc;
      float ktv = ktS[kc];
#pragma unroll
      for (int rr = 0; rr < 8; ++rr) {
        int rl = mt * 16 + rr + 8 * half;
        float w;
        if (key <= q0 + rl) {
          float val = sacc[jt][rr] - qtS[rl] * ktv;
          float ratio = fmaxf(-val / Kh, 1.0f + 1e-7f);
          w = -sqrtKh * __logf(ratio + sqrtf(ratio * ratio - 1.0f)); // acosh
        } else {
          w = -3.0e38f;
        }
        u.sp.S[rl][kc] = w;
      }
    }
    __syncthreads();

    // ---- online softmax row pass; writes P in A-frag layout ----
    if (tid < 64) {
      int r = tid;
      float mold = rowm[r];
      float bmax = -3.0e38f;
      for (int jj = 0; jj < 64; ++jj) bmax = fmaxf(bmax, u.sp.S[r][jj]);
      float mnew = fmaxf(mold, bmax);
      float scale = __expf(mold - mnew);
      float lsum = 0.f;
      for (int pj = 0; pj < 32; ++pj) {
        float p0 = __expf(u.sp.S[r][2 * pj] - mnew);
        float p1 = __expf(u.sp.S[r][2 * pj + 1] - mnew);
        lsum += p0 + p1;
        int ktl = pj >> 4, pp = pj & 15;
        int lt = ((pp >> 2) & 1) * 16 + (r & 15);
        int sl = ((pp >> 3) << 2) | (pp & 3);
        u.sp.Pf[r >> 4][ktl][lt * 8 + sl] = pk2(p0, p1);
      }
      rowl[r] = rowl[r] * scale + lsum;
      rowm[r] = mnew;
      rowscale[r] = scale;
    }
    __syncthreads();

    // ---- rescale accumulators, P @ LV via WMMA ----
    float scl[8];
#pragma unroll
    for (int rr = 0; rr < 8; ++rr) scl[rr] = rowscale[mt * 16 + rr + 8 * half];
#pragma unroll
    for (int rr = 0; rr < 8; ++rr) {
      o0[rr] *= scl[rr]; o1[rr] *= scl[rr]; o2[rr] *= scl[rr];
    }
#pragma unroll
    for (int ktl = 0; ktl < 2; ++ktl) {
      v16bf af = ldfrag(&u.sp.Pf[mt][ktl][lane * 8]);
      v16bf bb0 = ldfrag(&LVf[ktl][no0 + 0][lane * 8]);
      v16bf bb1 = ldfrag(&LVf[ktl][no0 + 1][lane * 8]);
      v16bf bb2 = ldfrag(&LVf[ktl][no0 + 2][lane * 8]);
      o0 = __builtin_amdgcn_wmma_f32_16x16x32_bf16(false, af, false, bb0,
                                                   (short)0, o0, false, false);
      o1 = __builtin_amdgcn_wmma_f32_16x16x32_bf16(false, af, false, bb1,
                                                   (short)0, o1, false, false);
      o2 = __builtin_amdgcn_wmma_f32_16x16x32_bf16(false, af, false, bb2,
                                                   (short)0, o2, false, false);
    }
    __syncthreads();
  }

  // ---- stage outputs, normalize per row, write ----
#pragma unroll
  for (int rr = 0; rr < 8; ++rr) {
    int rl = mt * 16 + rr + 8 * half;
    u.Ost[rl][(no0 + 0) * 16 + col] = o0[rr];
    u.Ost[rl][(no0 + 1) * 16 + col] = o1[rr];
    u.Ost[rl][(no0 + 2) * 16 + col] = o2[rr];
  }
  __syncthreads();
  if (tid < 64) {
    int r = tid;
    float inv_l = 1.0f / rowl[r];
    float v0 = u.Ost[r][0] * inv_l;
    float s2 = 0.f;
    for (int e = 1; e < 65; ++e) { float v = u.Ost[r][e] * inv_l; s2 += v * v; }
    float ln2 = v0 * v0 - s2;
    float hn = sqrtf(fmaxf(ln2, 1e-12f));
    float scl = sqrtKh / hn;
    float* orow = O + ((size_t)(b * TT + q0 + r)) * (HH * 65) + h * 65;
    orow[0] = v0 * scl;
    for (int e = 1; e < 65; ++e) orow[e] = (u.Ost[r][e] * inv_l) * scl;
  }
}

// ---------------------------------------------------------------------------
// Host-side orchestration
// ---------------------------------------------------------------------------
extern "C" void kernel_launch(void* const* d_in, const int* in_sizes, int n_in,
                              void* d_out, int out_size, void* d_ws, size_t ws_size,
                              hipStream_t stream) {
  (void)in_sizes; (void)n_in; (void)out_size; (void)ws_size;
  const float* x      = (const float*)d_in[0];
  const float* blk_c  = (const float*)d_in[1];
  const float* mlp_c  = (const float*)d_in[2];
  const float* attn_c = (const float*)d_in[3];
  const float* w_qkv  = (const float*)d_in[4];
  const float* w_ap   = (const float*)d_in[5];
  const float* b_ap   = (const float*)d_in[6];
  const float* w_me   = (const float*)d_in[7];
  const float* b_me   = (const float*)d_in[8];
  const float* w_ms   = (const float*)d_in[9];
  const float* b_ms   = (const float*)d_in[10];
  const float* ln_w   = (const float*)d_in[11];
  const float* ln_b   = (const float*)d_in[12];
  float* out = (float*)d_out;

  float* ws = (float*)d_ws;
  unsigned* wsu = (unsigned*)d_ws;
  const size_t M = MM;
  // fp32 buffers
  size_t o_qkv = 0;                                  // M x 3072
  size_t o_Qr  = o_qkv + M * 3072;
  size_t o_Kr  = o_Qr  + (size_t)BB * HH * TT * DD;
  size_t o_Vr  = o_Kr  + (size_t)BB * HH * TT * DD;
  size_t o_qt  = o_Vr  + (size_t)BB * HH * TT * DD;
  size_t o_kt  = o_qt  + (size_t)BB * HH * TT;
  size_t o_lv0 = o_kt  + (size_t)BB * HH * TT;
  size_t o_O   = o_lv0 + (size_t)BB * HH * TT;       // M x 1040
  size_t o_x2  = o_O   + M * 1040;                   // M x 1024
  size_t o_x3  = o_x2  + M * 1024;                   // M x 1024
  size_t o_g   = o_qkv;   // M x 4099, alias: qkv/Qr (and Kr head) dead by then
  // swizzled activations (u32 tiles, [kt][mt=256][256])
  size_t o_lz  = o_x3 + M * 1024;                  // ln1/ln2: KT=33
  size_t o_hz  = o_lz + (size_t)33 * 256 * 256;    // H: KT=129
  // swizzled weights (u32 tiles, [kt][NT][256])
  size_t o_qz  = o_hz + (size_t)129 * 256 * 256;   // qkv: KT=33,  NT=192
  size_t o_az  = o_qz + (size_t)33 * 192 * 256;    // ap:  KT=33,  NT=64
  size_t o_ez  = o_az + (size_t)33 * 64 * 256;     // me:  KT=33,  NT=264
  size_t o_sz  = o_ez + (size_t)33 * 264 * 256;    // ms:  KT=129, NT=64

  dim3 blk(256);

  // 0) weight pre-swizzle (fragment-layout bf16 pair tiles)
  wswz_kernel<<<dim3(33 * 192), blk, 0, stream>>>(w_qkv, wsu + o_qz, 1025, 3072, 192, 33);
  wswz_kernel<<<dim3(33 * 64),  blk, 0, stream>>>(w_ap,  wsu + o_az, 1040, 1024, 64, 33);
  wswz_kernel<<<dim3(33 * 264), blk, 0, stream>>>(w_me,  wsu + o_ez, 1025, 4099, 264, 33);
  wswz_kernel<<<dim3(129 * 64), blk, 0, stream>>>(w_ms,  wsu + o_sz, 4100, 1024, 64, 129);

  // 1) ln1 = project(layer_norm(x)) -> swizzled A tiles
  row_project_kernel<<<dim3((unsigned)M), blk, 0, stream>>>(
      x, nullptr, wsu + o_lz, ln_w, ln_b, blk_c, EE, 1);

  // 2) qkv = ln1 @ w_qkv   (both operands async block copies)
  gemm_wmma_kernel<<<dim3(3072 / 128, M / 128), blk, 0, stream>>>(
      nullptr, wsu + o_lz, wsu + o_qz, nullptr, nullptr,
      ws + o_qkv, (int)M, 3072, 1025, 192);

  // 3) RoPE + Lorentz time components
  rope_kernel<<<dim3((BB * HH * TT) / 4), blk, 0, stream>>>(
      ws + o_qkv, attn_c, ws + o_Qr, ws + o_Kr, ws + o_Vr,
      ws + o_qt, ws + o_kt, ws + o_lv0);

  // 4) hyperbolic flash attention -> O [M,1040]
  attn_kernel<<<dim3(TT / 64, BB * HH), blk, 0, stream>>>(
      ws + o_Qr, ws + o_Kr, ws + o_Vr, ws + o_qt, ws + o_kt, ws + o_lv0,
      attn_c, ws + o_O);

  // 5) x2 = x + O @ w_attn_proj + b   (A converted in-kernel)
  gemm_wmma_kernel<<<dim3(1024 / 128, M / 128), blk, 0, stream>>>(
      ws + o_O, nullptr, wsu + o_az, b_ap, x,
      ws + o_x2, (int)M, 1024, 1040, 64);

  // 6) ln2 = project(layer_norm(x2)) -> swizzled A tiles
  row_project_kernel<<<dim3((unsigned)M), blk, 0, stream>>>(
      ws + o_x2, nullptr, wsu + o_lz, ln_w, ln_b, blk_c, EE, 1);

  // 7) g = ln2 @ w_mlp_expand + b
  gemm_wmma_kernel<<<dim3((4099 + 127) / 128, M / 128), blk, 0, stream>>>(
      nullptr, wsu + o_lz, wsu + o_ez, b_me, nullptr,
      ws + o_g, (int)M, 4099, 1025, 264);

  // 8) H = project(gelu(g), mlp_curv) -> swizzled A tiles
  gelu_project_kernel<<<dim3((unsigned)M), blk, 0, stream>>>(
      ws + o_g, wsu + o_hz, mlp_c, 4099);

  // 9) x3 = x2 + H @ w_mlp_shrink + b
  gemm_wmma_kernel<<<dim3(1024 / 128, M / 128), blk, 0, stream>>>(
      nullptr, wsu + o_hz, wsu + o_sz, b_ms, ws + o_x2,
      ws + o_x3, (int)M, 1024, 4100, 64);

  // 10) out = project(x3)
  row_project_kernel<<<dim3((unsigned)M), blk, 0, stream>>>(
      ws + o_x3, out, nullptr, nullptr, nullptr, blk_c, EE, 0);
}
